// QuantumGate_65481071404610
// MI455X (gfx1250) — compile-verified, matrix operand order checked by reading, not run on hardware
//
#include <hip/hip_runtime.h>

// Quantum circuit <Z_q> — algebraically reduced.
// state = RX-product-state; all later gates are diagonal (RZ) or basis
// permutations (CNOT ladder = prefix-xor over GF(2)), so probabilities are a
// permuted product distribution and  <Z_q> = prod_{i<=q, (q-i)%4==0} cos(x_i).
// Evaluated as a 24x24 {0,1}-mask mat-vec in log-domain via WMMA f32 16x16x4,
// signs handled by parity of negative cosines. Single wave32, ~launch-bound
// (vs ~56 GB of HBM traffic / ~2.4 ms for the dense statevector reference).

#define NQ 24

typedef __attribute__((ext_vector_type(2)))  float    v2f;
typedef __attribute__((ext_vector_type(8)))  float    v8f;
typedef __attribute__((ext_vector_type(16))) _Float16 v16h;

// Row q of M = A^4 (A = CNOT-ladder prefix-xor matrix over GF(2)):
// ones exactly at i = q, q-4, q-8, ...
__device__ __forceinline__ float mask_qi(int q, int i) {
  return (i <= q && (((q - i) & 3) == 0)) ? 1.0f : 0.0f;
}

__global__ __launch_bounds__(32)
void qz24_kernel(const float* __restrict__ x, float* __restrict__ out) {
  __shared__ float s_log[NQ];
  __shared__ float s_cos[NQ];
  __shared__ float s_sum[32];

  const int lane = (int)threadIdx.x;

  if (lane < NQ) {
    const float c = cosf(x[lane]);
    s_cos[lane] = c;
    // Clamp so a (theoretical) cos==0 gives log(FLT_MIN) ~ -87.3 instead of
    // -inf: avoids 0 * -inf = NaN inside the WMMA dot products; exp() of the
    // clamped sum still underflows to ~0, matching the true product.
    s_log[lane] = logf(fmaxf(fabsf(c), 1.1754944e-38f));
  }
  __syncthreads();

  const int m = lane & 15;   // tile row (A/D) / column n (B/D)
  const int h = lane >> 4;   // lane-half selects K sub-block

#if __has_builtin(__builtin_amdgcn_wmma_f32_16x16x4_f32)
  // D(16x16 f32) = A(16x4 f32) x B(4x16 f32) + C, 6 K-chunks cover K=24.
  // A layout (ISA 7.12.2): lanes 0-15 VGPR0/1 = K0/K1, lanes 16-31 = K2/K3.
  // B assumed symmetric: lanes = N, VGPR0/1 = K pair per lane-half; only
  // column n==0 carries the log-vector.
  for (int t = 0; t < 2; ++t) {
    const int rb = t * 16;   // global row base of this tile (rows 16..23 used in t=1)
    v8f acc = {};
#pragma unroll
    for (int kk = 0; kk < 6; ++kk) {
      const int k0 = kk * 4 + h * 2;
      v2f a, b;
      a.x = mask_qi(rb + m, k0);
      a.y = mask_qi(rb + m, k0 + 1);
      const bool n0 = (m == 0);
      b.x = n0 ? s_log[k0]     : 0.0f;
      b.y = n0 ? s_log[k0 + 1] : 0.0f;
      acc = __builtin_amdgcn_wmma_f32_16x16x4_f32(false, a, false, b,
                                                  (short)0, acc, false, false);
    }
    // D layout: VGPR r holds row r (lanes 0-15, n=lane) / row r+8 (lanes 16-31).
    // Column n=0 lives in lane 0 (rows rb+0..7) and lane 16 (rows rb+8..15).
    if (lane == 0) {
#pragma unroll
      for (int r = 0; r < 8; ++r) {
        const int row = rb + r;
        if (row < NQ) s_sum[row] = acc[r];
      }
    } else if (lane == 16) {
#pragma unroll
      for (int r = 0; r < 8; ++r) {
        const int row = rb + 8 + r;
        if (row < NQ) s_sum[row] = acc[r];
      }
    }
  }
#else
  // Fallback: codegen-confirmed f16 WMMA, whole K=24 (zero-padded to 32) in
  // one shot per row-tile. Mask entries are exact in f16; logs lose ~2^-11.
  for (int t = 0; t < 2; ++t) {
    const int rb = t * 16;
    v16h a, b;
#pragma unroll
    for (int e = 0; e < 16; ++e) {
      // 16-bit A 16x32 layout: VGPR0-3 = K (h*8 .. h*8+7), VGPR4-7 = +16.
      const int ka = (e < 8) ? (h * 8 + e) : (8 + h * 8 + e);
      a[e] = (_Float16)mask_qi(rb + m, ka);
      // B 32x16 packed in contiguous K order (unlike A): lanes 0-15 hold K=0..15, lanes 16-31 K=16..31.
      const int kb = h * 16 + e;
      const float bv = (m == 0 && kb < NQ) ? s_log[kb] : 0.0f;
      b[e] = (_Float16)bv;
    }
    v8f acc = {};
    acc = __builtin_amdgcn_wmma_f32_16x16x32_f16(false, a, false, b,
                                                 (short)0, acc, false, false);
    if (lane == 0) {
#pragma unroll
      for (int r = 0; r < 8; ++r) {
        const int row = rb + r;
        if (row < NQ) s_sum[row] = acc[r];
      }
    } else if (lane == 16) {
#pragma unroll
      for (int r = 0; r < 8; ++r) {
        const int row = rb + 8 + r;
        if (row < NQ) s_sum[row] = acc[r];
      }
    }
  }
#endif
  __syncthreads();

  if (lane < NQ) {
    // Sign = parity of negative cosines in the mask row.
    float sign = 1.0f;
    for (int i = lane; i >= 0; i -= 4)
      if (s_cos[i] < 0.0f) sign = -sign;
    out[lane] = sign * expf(s_sum[lane]);
  }
}

extern "C" void kernel_launch(void* const* d_in, const int* in_sizes, int n_in,
                              void* d_out, int out_size, void* d_ws, size_t ws_size,
                              hipStream_t stream) {
  (void)in_sizes; (void)n_in; (void)out_size; (void)d_ws; (void)ws_size;
  const float* x = (const float*)d_in[0];
  float* out = (float*)d_out;
  qz24_kernel<<<1, 32, 0, stream>>>(x, out);
}
